// RNNDecoderNoAttn_88081189307133
// MI455X (gfx1250) — matrix operand order repaired and checked by reading, not compiled, not run on hardware
//
#include <hip/hip_runtime.h>

// ---------------------------------------------------------------------------
// Persistent LSTM-decoder kernel for MI455X (gfx1250, wave32, WMMA).
//
//   B=1024 batch rows, H=128 hidden, T=512 steps, gates=4H=512, MID=64.
//   64 workgroups x 128 threads (4 waves). Each workgroup owns 16 batch rows
//   for the whole sequence; W_hh/W1 cached in LDS as f16; c-state in VGPRs.
//   Per step, per wave: 32 WMMA (gates) + 4 WMMA (MLP head), f32 accumulate.
// ---------------------------------------------------------------------------

typedef __attribute__((ext_vector_type(16))) _Float16 v16h;
typedef __attribute__((ext_vector_type(8)))  _Float16 v8h;
typedef __attribute__((ext_vector_type(8)))  float    v8f;

#define BATCH 1024
#define HID   128
#define TLEN  512
#define GATES 512   // 4*H
#define MIDN  64    // H/2
#define MT    16    // batch rows per workgroup

__device__ __forceinline__ float sigf(float x) {
    return 1.0f / (1.0f + __expf(-x));
}

// A-fragment (16x32 f16): lane<16 -> rows m=lane, K {0..15};
// lane>=16 -> K {16..31}. Two 16B contiguous LDS loads per fragment.
__device__ __forceinline__ v16h load_a_frag(const _Float16* sH, int k0, int lane) {
    const int m  = lane & 15;
    const int hi = lane >> 4;
    const v8h lo  = *(const v8h*)(sH + m * HID + k0 + 16 * hi);
    const v8h hiv = *(const v8h*)(sH + m * HID + k0 + 16 * hi + 8);
    v16h a;
#pragma unroll
    for (int i = 0; i < 8; ++i) { a[i] = lo[i]; a[i + 8] = hiv[i]; }
    return a;
}

// B-fragment (32x16 f16): lane n=lane&15 holds K = k0 + 16*(lane>=16) .. +15,
// i.e. 16 contiguous halves (one 32B LDS load) from weight stored [N][K].
__device__ __forceinline__ v16h load_b_frag(const _Float16* sW, int n0, int k0,
                                            int lane, int kstride) {
    const int n  = lane & 15;
    const int kh = (lane >> 4) * 16;
    return *(const v16h*)(sW + (n0 + n) * kstride + k0 + kh);
}

__global__ __launch_bounds__(128) void
RNNDecoderNoAttn_88081189307133_kernel(const float* __restrict__ h0,
                                       const float* __restrict__ c0,
                                       const float* __restrict__ Wih,
                                       const float* __restrict__ Whh,
                                       const float* __restrict__ bih,
                                       const float* __restrict__ bhh,
                                       const float* __restrict__ W1,
                                       const float* __restrict__ b1,
                                       const float* __restrict__ W2,
                                       const float* __restrict__ b2,
                                       float* __restrict__ out)
{
    extern __shared__ char smem[];
    _Float16* sWhh = (_Float16*)smem;            // [512][128] f16  (128 KB)
    _Float16* sW1  = sWhh + GATES * HID;         // [ 64][128] f16  ( 16 KB)
    _Float16* sH   = sW1  + MIDN * HID;          // [ 16][128] f16  (  4 KB)
    float* sZ    = (float*)(sH + MT * HID);      // [16][64] f32    (  4 KB)
    float* sBias = sZ + MT * MIDN;               // [512]
    float* sWih  = sBias + GATES;                // [512]
    float* sB1   = sWih + GATES;                 // [64]
    float* sW2   = sB1 + MIDN;                   // [64]
    float* sY    = sW2 + MIDN;                   // [16]

    const int tid  = threadIdx.x;
    const int lane = tid & 31;
    const int w    = tid >> 5;          // wave id 0..3 -> hidden slice [32w,32w+32)
    const int hi   = lane >> 4;         // row-half selector for C/D tiles
    const int nl   = lane & 15;         // column within a 16-wide tile
    const int row0 = blockIdx.x * MT;   // first batch row of this workgroup

    // ---- cooperative LDS fill (weights converted fp32 -> f16 once) ----
    for (int i = tid; i < GATES * HID; i += 128) sWhh[i] = (_Float16)Whh[i];
    for (int i = tid; i < MIDN * HID;  i += 128) sW1[i]  = (_Float16)W1[i];
    for (int i = tid; i < GATES; i += 128) { sBias[i] = bih[i] + bhh[i]; sWih[i] = Wih[i]; }
    for (int i = tid; i < MIDN;  i += 128) { sB1[i] = b1[i]; sW2[i] = W2[i]; }
    for (int i = tid; i < MT * HID; i += 128) sH[i] = (_Float16)h0[row0 * HID + i];
    if (tid < MT) sY[tid] = 0.0f;       // first-step decoder input is zero

    // ---- c-state lives in registers: two 16x16 f32 tiles per wave ----
    v8f creg[2];
#pragma unroll
    for (int s = 0; s < 2; ++s) {
        const int j = 32 * w + 16 * s + nl;
#pragma unroll
        for (int r = 0; r < 8; ++r)
            creg[s][r] = c0[(row0 + r + 8 * hi) * HID + j];
    }

    __syncthreads();
    const float b2v = *b2;

    for (int t = 0; t < TLEN; ++t) {
        // ---- A fragments of current h (shared across all N tiles) ----
        v16h a[4];
#pragma unroll
        for (int kc = 0; kc < 4; ++kc) a[kc] = load_a_frag(sH, kc * 32, lane);

        float ym[8];
#pragma unroll
        for (int r = 0; r < 8; ++r) ym[r] = sY[r + 8 * hi];

        // ---- gates = h @ Whh^T + (b_ih+b_hh) + y_prev * W_ih  (rank-1) ----
        v8f acc[4][2];
#pragma unroll
        for (int g = 0; g < 4; ++g) {
#pragma unroll
            for (int s = 0; s < 2; ++s) {
                const int n0 = 128 * g + 32 * w + 16 * s;
                const float bn = sBias[n0 + nl];
                const float wn = sWih[n0 + nl];
                v8f c;
#pragma unroll
                for (int r = 0; r < 8; ++r) c[r] = bn + ym[r] * wn;
#pragma unroll
                for (int kc = 0; kc < 4; ++kc) {
                    const v16h bf = load_b_frag(sWhh, n0, kc * 32, lane, HID);
                    c = __builtin_amdgcn_wmma_f32_16x16x32_f16(
                            false, a[kc], false, bf, (short)0, c, false, false);
                }
                acc[g][s] = c;
            }
        }

        __syncthreads();   // all waves done reading old h before we overwrite it

        // ---- fused gate nonlinearities + c/h update; new h -> LDS ----
#pragma unroll
        for (int s = 0; s < 2; ++s) {
            const int j = 32 * w + 16 * s + nl;
#pragma unroll
            for (int r = 0; r < 8; ++r) {
                const float iv = sigf(acc[0][s][r]);
                const float fv = sigf(acc[1][s][r]);
                const float gv = tanhf(acc[2][s][r]);
                const float ov = sigf(acc[3][s][r]);
                const float cc = fv * creg[s][r] + iv * gv;
                creg[s][r] = cc;
                sH[(r + 8 * hi) * HID + j] = (_Float16)(ov * tanhf(cc));
            }
        }
        __syncthreads();

        // ---- head: z = relu(h @ W1^T + b1); each wave one 16x16 slab ----
        {
            v16h ah[4];
#pragma unroll
            for (int kc = 0; kc < 4; ++kc) ah[kc] = load_a_frag(sH, kc * 32, lane);
            const int n0 = 16 * w;
            const float bz = sB1[n0 + nl];
            v8f z;
#pragma unroll
            for (int r = 0; r < 8; ++r) z[r] = bz;
#pragma unroll
            for (int kc = 0; kc < 4; ++kc) {
                const v16h bf = load_b_frag(sW1, n0, kc * 32, lane, HID);
                z = __builtin_amdgcn_wmma_f32_16x16x32_f16(
                        false, ah[kc], false, bf, (short)0, z, false, false);
            }
#pragma unroll
            for (int r = 0; r < 8; ++r)
                sZ[(r + 8 * hi) * MIDN + n0 + nl] = fmaxf(z[r], 0.0f);
        }
        __syncthreads();

        // ---- y = sigmoid(z @ W2^T + b2): 16 rows, one lane each ----
        if (tid < MT) {
            float a2 = b2v;
#pragma unroll
            for (int jj = 0; jj < MIDN; ++jj) a2 += sZ[tid * MIDN + jj] * sW2[jj];
            const float yv = sigf(a2);
            sY[tid] = yv;
            out[(row0 + tid) * TLEN + t] = yv;   // output already (B, T, 1)
        }
        __syncthreads();
    }
}

extern "C" void kernel_launch(void* const* d_in, const int* in_sizes, int n_in,
                              void* d_out, int out_size, void* d_ws, size_t ws_size,
                              hipStream_t stream) {
    (void)in_sizes; (void)n_in; (void)d_ws; (void)ws_size; (void)out_size;
    const float* h0  = (const float*)d_in[0];
    const float* c0  = (const float*)d_in[1];
    // d_in[2] encoder_outputs: unused by the reference computation
    const float* Wih = (const float*)d_in[3];
    const float* Whh = (const float*)d_in[4];
    const float* bih = (const float*)d_in[5];
    const float* bhh = (const float*)d_in[6];
    const float* W1  = (const float*)d_in[7];
    const float* b1  = (const float*)d_in[8];
    const float* W2  = (const float*)d_in[9];
    const float* b2  = (const float*)d_in[10];
    float* out = (float*)d_out;

    const size_t shmem =
        (size_t)(GATES * HID + MIDN * HID + MT * HID) * sizeof(_Float16) +
        (size_t)(MT * MIDN + GATES + GATES + MIDN + MIDN + MT) * sizeof(float);

    dim3 grid(BATCH / MT), block(128);
    hipLaunchKernelGGL(RNNDecoderNoAttn_88081189307133_kernel, grid, block, shmem, stream,
                       h0, c0, Wih, Whh, bih, bhh, W1, b1, W2, b2, out);
}
